// GraphStateEncoder_51823075393574
// MI455X (gfx1250) — matrix operand order repaired and checked by reading, not compiled, not run on hardware
//
#include <hip/hip_runtime.h>

// ---------------------------------------------------------------------------
// CDNA5 wave32 WMMA types
// ---------------------------------------------------------------------------
typedef __attribute__((ext_vector_type(16))) __bf16 v16bf;
typedef __attribute__((ext_vector_type(8)))  float  v8f;

union FragB {
    v16bf bf;
    unsigned int u[8];
    unsigned short s[16];
};

__device__ __forceinline__ unsigned short f2bf(float f) {
    unsigned int u = __float_as_uint(f);
    unsigned int r = u + 0x7FFFu + ((u >> 16) & 1u);   // round-to-nearest-even
    return (unsigned short)(r >> 16);
}

// ---------------------------------------------------------------------------
// Pack a row-major f32 weight W[K][NCOL] into wave32 bf16 B-fragments.
// Fragment (nt, kt): lane n<16 holds column nt*16+n, K = kt*32 + {0..7,16..23} (16 bf16,
// 2 per VGPR); lane n+16 holds K = kt*32 + {8..15,24..31}.  K beyond the
// logical K is zero-padded.
// ---------------------------------------------------------------------------
__global__ void pack_b_kernel(const float* __restrict__ W,
                              unsigned short* __restrict__ out,
                              int K, int NCOL, int KT) {
    int t = blockIdx.x * blockDim.x + threadIdx.x;
    int NT = NCOL >> 4;
    int total = NT * KT * 32;
    if (t >= total) return;
    int lane = t & 31;
    int kt   = (t >> 5) % KT;
    int nt   = t / (32 * KT);
    int col   = nt * 16 + (lane & 15);
    int kbase = kt * 32 + ((lane & 16) ? 8 : 0);
    unsigned short* o = out + ((size_t)((nt * KT + kt) * 32 + lane)) * 16;
#pragma unroll
    for (int j = 0; j < 16; ++j) {
        int k = kbase + j + ((j >= 8) ? 8 : 0);
        float v = (k < K) ? W[k * NCOL + col] : 0.0f;
        o[j] = f2bf(v);
    }
}

// ---------------------------------------------------------------------------
// Input feature gather: x[row, k] for the 38-wide concatenated input
// ---------------------------------------------------------------------------
__device__ __forceinline__ float xval(int row, int k,
                                      const float* __restrict__ bfeat,
                                      const int*   __restrict__ rid,
                                      const float* __restrict__ remb,
                                      const float* __restrict__ iemb) {
    if (k < 14) return bfeat[row * 14 + k];
    if (k < 30) return remb[rid[row] * 16 + (k - 14)];
    if (k < 38) return iemb[(row & 1023) * 8 + (k - 30)];
    return 0.0f;
}

// Build A-fragment (16x32 bf16) from row-major bf16 LDS tile [16][128].
// Lane l<16: row l, elems 0..7 = K c0..c0+7, elems 8..15 = K c0+16..c0+23
// with c0 = kt*32 + (lane>=16 ? 8 : 0).
__device__ __forceinline__ FragB load_a_frag(const unsigned short* lds,
                                             int lane, int kt) {
    int l = lane & 15, half = lane >> 4;
    int base = l * 128 + kt * 32 + half * 8;       // multiple of 8 elems = 16B
    const uint4* p = (const uint4*)lds;
    uint4 q0 = p[base >> 3];
    uint4 q1 = p[(base + 16) >> 3];
    FragB fr;
    fr.u[0] = q0.x; fr.u[1] = q0.y; fr.u[2] = q0.z; fr.u[3] = q0.w;
    fr.u[4] = q1.x; fr.u[5] = q1.y; fr.u[6] = q1.z; fr.u[7] = q1.w;
    return fr;
}

// ---------------------------------------------------------------------------
// Kernel 1: emb = relu(relu(x @ W1 + b1) @ W2 + b2)     (one wave per 16 rows)
// ---------------------------------------------------------------------------
__global__ __launch_bounds__(32)
void emb_kernel(const float* __restrict__ bfeat, const int* __restrict__ rid,
                const float* __restrict__ remb, const float* __restrict__ iemb,
                const unsigned short* __restrict__ W1p, const float* __restrict__ b1,
                const unsigned short* __restrict__ W2p, const float* __restrict__ b2,
                float* __restrict__ emb, int N) {
    int tile = blockIdx.x;
    int lane = threadIdx.x;
    int l = lane & 15, half = lane >> 4;
    int row = tile * 16 + l;
    int rowc = (row < N) ? row : (N - 1);

    __shared__ __align__(16) unsigned short hl[16 * 128];

    // A fragments of x (K = 38 padded to 64 -> 2 k-tiles)
    FragB a0, a1;
    int c0 = half * 8, c1 = 16 + half * 8;
#pragma unroll
    for (int j = 0; j < 8; ++j) {
        a0.s[j]     = f2bf(xval(rowc, c0 + j,      bfeat, rid, remb, iemb));
        a0.s[8 + j] = f2bf(xval(rowc, c1 + j,      bfeat, rid, remb, iemb));
        a1.s[j]     = f2bf(xval(rowc, 32 + c0 + j, bfeat, rid, remb, iemb));
        a1.s[8 + j] = f2bf(xval(rowc, 32 + c1 + j, bfeat, rid, remb, iemb));
    }

    const v16bf* W1f = (const v16bf*)W1p;
#pragma unroll
    for (int nt = 0; nt < 8; ++nt) {
        v8f acc = {0.f, 0.f, 0.f, 0.f, 0.f, 0.f, 0.f, 0.f};
        v16bf bB0 = W1f[(nt * 2 + 0) * 32 + lane];
        v16bf bB1 = W1f[(nt * 2 + 1) * 32 + lane];
        acc = __builtin_amdgcn_wmma_f32_16x16x32_bf16(false, a0.bf, false, bB0,
                                                      (short)0, acc, false, false);
        acc = __builtin_amdgcn_wmma_f32_16x16x32_bf16(false, a1.bf, false, bB1,
                                                      (short)0, acc, false, false);
        float bias = b1[nt * 16 + l];
#pragma unroll
        for (int v = 0; v < 8; ++v) {
            float h = acc[v] + bias;
            h = h > 0.f ? h : 0.f;
            hl[(v + half * 8) * 128 + nt * 16 + l] = f2bf(h);
        }
    }
    __syncthreads();

    FragB ha[4];
#pragma unroll
    for (int kt = 0; kt < 4; ++kt) ha[kt] = load_a_frag(hl, lane, kt);

    const v16bf* W2f = (const v16bf*)W2p;
#pragma unroll
    for (int nt = 0; nt < 8; ++nt) {
        v8f acc = {0.f, 0.f, 0.f, 0.f, 0.f, 0.f, 0.f, 0.f};
#pragma unroll
        for (int kt = 0; kt < 4; ++kt) {
            v16bf bB = W2f[(nt * 4 + kt) * 32 + lane];
            acc = __builtin_amdgcn_wmma_f32_16x16x32_bf16(false, ha[kt].bf, false, bB,
                                                          (short)0, acc, false, false);
        }
        float bias = b2[nt * 16 + l];
#pragma unroll
        for (int v = 0; v < 8; ++v) {
            int r = tile * 16 + v + half * 8;
            float e = acc[v] + bias;
            e = e > 0.f ? e : 0.f;
            if (r < N) emb[r * 128 + nt * 16 + l] = e;
        }
    }
}

// ---------------------------------------------------------------------------
// Kernel 2: pin-weight scatter  pw[blk] += w
// ---------------------------------------------------------------------------
__global__ void pw_kernel(const int* __restrict__ blk, const float* __restrict__ w,
                          float* __restrict__ pw, int P) {
    int t = blockIdx.x * blockDim.x + threadIdx.x;
    if (t < P) atomicAdd(&pw[blk[t]], w[t]);
}

// ---------------------------------------------------------------------------
// Kernel 3: symmetric edge scatter (one wave / edge; lane owns 4 columns).
// emb & agg are L2-resident (51.2MB each << 192MB), atomics resolve in L2.
// ---------------------------------------------------------------------------
__global__ void b2b_kernel(const int* __restrict__ src, const int* __restrict__ dst,
                           const float* __restrict__ w, const float* __restrict__ emb,
                           float* __restrict__ agg, int E) {
    int t = blockIdx.x * blockDim.x + threadIdx.x;
    int e = t >> 5;
    if (e >= E) return;
    int lane = t & 31;
    int s = src[e], d = dst[e];
    float ww = w[e];
    const float4* emb4 = (const float4*)emb;
    float4 es = emb4[s * 32 + lane];
    float4 ed = emb4[d * 32 + lane];
    float* as_ = agg + s * 128 + lane * 4;
    float* ad_ = agg + d * 128 + lane * 4;
    atomicAdd(as_ + 0, ed.x * ww); atomicAdd(as_ + 1, ed.y * ww);
    atomicAdd(as_ + 2, ed.z * ww); atomicAdd(as_ + 3, ed.w * ww);
    atomicAdd(ad_ + 0, es.x * ww); atomicAdd(ad_ + 1, es.y * ww);
    atomicAdd(ad_ + 2, es.z * ww); atomicAdd(ad_ + 3, es.w * ww);
}

// ---------------------------------------------------------------------------
// Kernel 4: a = agg + emb*pw ; m = relu(a@M1+mb1)@M2+mb2 ;
//           out = LayerNorm(emb + m) ; gsum += column sums of out
// ---------------------------------------------------------------------------
__global__ __launch_bounds__(32)
void msg_kernel(const float* __restrict__ emb, const float* __restrict__ agg,
                const float* __restrict__ pw,
                const unsigned short* __restrict__ M1p, const float* __restrict__ mb1,
                const unsigned short* __restrict__ M2p, const float* __restrict__ mb2,
                const float* __restrict__ ln_g, const float* __restrict__ ln_b,
                float* __restrict__ out, float* __restrict__ gsum, int N) {
    int tile = blockIdx.x;
    int lane = threadIdx.x;
    int l = lane & 15, half = lane >> 4;

    __shared__ __align__(16) unsigned short al[16 * 128];
    __shared__ __align__(16) unsigned short hl[16 * 128];
    __shared__ __align__(16) float fl[16 * 128];

    // Stage a = agg + emb*pw as bf16 tile [16][128]
    const float4* emb4 = (const float4*)emb;
    const float4* agg4 = (const float4*)agg;
    unsigned int* alu = (unsigned int*)al;
#pragma unroll
    for (int it = 0; it < 16; ++it) {   // row = it, float4-col = lane
        int rg = tile * 16 + it;
        int rgc = (rg < N) ? rg : (N - 1);
        float p = pw[rgc];
        float4 a = agg4[rgc * 32 + lane];
        float4 e = emb4[rgc * 32 + lane];
        float x0 = a.x + e.x * p, x1 = a.y + e.y * p;
        float x2 = a.z + e.z * p, x3 = a.w + e.w * p;
        unsigned int lo = (unsigned)f2bf(x0) | ((unsigned)f2bf(x1) << 16);
        unsigned int hi = (unsigned)f2bf(x2) | ((unsigned)f2bf(x3) << 16);
        int f = it * 32 + lane;
        alu[f * 2 + 0] = lo;
        alu[f * 2 + 1] = hi;
    }
    __syncthreads();

    FragB aa[4];
#pragma unroll
    for (int kt = 0; kt < 4; ++kt) aa[kt] = load_a_frag(al, lane, kt);

    const v16bf* M1f = (const v16bf*)M1p;
#pragma unroll
    for (int nt = 0; nt < 8; ++nt) {
        v8f acc = {0.f, 0.f, 0.f, 0.f, 0.f, 0.f, 0.f, 0.f};
#pragma unroll
        for (int kt = 0; kt < 4; ++kt) {
            v16bf bB = M1f[(nt * 4 + kt) * 32 + lane];
            acc = __builtin_amdgcn_wmma_f32_16x16x32_bf16(false, aa[kt].bf, false, bB,
                                                          (short)0, acc, false, false);
        }
        float bias = mb1[nt * 16 + l];
#pragma unroll
        for (int v = 0; v < 8; ++v) {
            float h = acc[v] + bias;
            h = h > 0.f ? h : 0.f;
            hl[(v + half * 8) * 128 + nt * 16 + l] = f2bf(h);
        }
    }
    __syncthreads();

    FragB hh[4];
#pragma unroll
    for (int kt = 0; kt < 4; ++kt) hh[kt] = load_a_frag(hl, lane, kt);

    const v16bf* M2f = (const v16bf*)M2p;
#pragma unroll
    for (int nt = 0; nt < 8; ++nt) {
        v8f acc = {0.f, 0.f, 0.f, 0.f, 0.f, 0.f, 0.f, 0.f};
#pragma unroll
        for (int kt = 0; kt < 4; ++kt) {
            v16bf bB = M2f[(nt * 4 + kt) * 32 + lane];
            acc = __builtin_amdgcn_wmma_f32_16x16x32_bf16(false, hh[kt].bf, false, bB,
                                                          (short)0, acc, false, false);
        }
        float bias = mb2[nt * 16 + l];
#pragma unroll
        for (int v = 0; v < 8; ++v) {
            int r = tile * 16 + v + half * 8;
            int rc = (r < N) ? r : (N - 1);
            float val = acc[v] + bias + emb[rc * 128 + nt * 16 + l];
            fl[(v + half * 8) * 128 + nt * 16 + l] = val;
        }
    }
    __syncthreads();

    // LayerNorm: lanes 0..15 each own one row
    if (lane < 16) {
        int r = tile * 16 + lane;
        const float4* flr = (const float4*)(fl + lane * 128);
        float s = 0.f, ss = 0.f;
#pragma unroll
        for (int i = 0; i < 32; ++i) {
            float4 v = flr[i];
            s  += v.x + v.y + v.z + v.w;
            ss += v.x * v.x + v.y * v.y + v.z * v.z + v.w * v.w;
        }
        float mu = s * (1.0f / 128.0f);
        float var = ss * (1.0f / 128.0f) - mu * mu;
        float rstd = rsqrtf(var + 1e-5f);
        const float4* g4 = (const float4*)ln_g;
        const float4* b4 = (const float4*)ln_b;
        float4* o4 = (float4*)(out + (size_t)r * 128);
        float4* fw = (float4*)(fl + lane * 128);
#pragma unroll
        for (int i = 0; i < 32; ++i) {
            float4 v = flr[i], g = g4[i], bb = b4[i];
            float4 y;
            y.x = (v.x - mu) * rstd * g.x + bb.x;
            y.y = (v.y - mu) * rstd * g.y + bb.y;
            y.z = (v.z - mu) * rstd * g.z + bb.z;
            y.w = (v.w - mu) * rstd * g.w + bb.w;
            if (r < N) { o4[i] = y; fw[i] = y; }
            else       { fw[i] = make_float4(0.f, 0.f, 0.f, 0.f); }
        }
    }
    __syncthreads();

    // Column sums for graph embedding: lane owns 4 columns
    const float4* flc = (const float4*)fl;
    float4 cs = make_float4(0.f, 0.f, 0.f, 0.f);
#pragma unroll
    for (int r = 0; r < 16; ++r) {
        float4 v = flc[r * 32 + lane];
        cs.x += v.x; cs.y += v.y; cs.z += v.z; cs.w += v.w;
    }
    atomicAdd(&gsum[lane * 4 + 0], cs.x);
    atomicAdd(&gsum[lane * 4 + 1], cs.y);
    atomicAdd(&gsum[lane * 4 + 2], cs.z);
    atomicAdd(&gsum[lane * 4 + 3], cs.w);
}

// ---------------------------------------------------------------------------
// Kernel 5: graph embedding = colsum / N
// ---------------------------------------------------------------------------
__global__ void fin_kernel(const float* __restrict__ gsum, float* __restrict__ out,
                           int N) {
    int c = threadIdx.x;
    out[(size_t)N * 128 + c] = gsum[c] / (float)N;
}

// ---------------------------------------------------------------------------
extern "C" void kernel_launch(void* const* d_in, const int* in_sizes, int n_in,
                              void* d_out, int out_size, void* d_ws, size_t ws_size,
                              hipStream_t stream) {
    const float* bfeat   = (const float*)d_in[0];
    const int*   rid     = (const int*)  d_in[1];
    const int*   b2b_src = (const int*)  d_in[2];
    const int*   b2b_dst = (const int*)  d_in[3];
    const float* b2b_w   = (const float*)d_in[4];
    const int*   p2b_blk = (const int*)  d_in[5];
    const float* p2b_w   = (const float*)d_in[6];
    const float* remb    = (const float*)d_in[7];
    const float* iemb    = (const float*)d_in[8];
    const float* W1      = (const float*)d_in[9];
    const float* b1      = (const float*)d_in[10];
    const float* W2      = (const float*)d_in[11];
    const float* b2      = (const float*)d_in[12];
    const float* M1      = (const float*)d_in[13];
    const float* mb1     = (const float*)d_in[14];
    const float* M2      = (const float*)d_in[15];
    const float* mb2     = (const float*)d_in[16];
    const float* ln_g    = (const float*)d_in[17];
    const float* ln_b    = (const float*)d_in[18];

    int N = in_sizes[1];
    int E = in_sizes[2];
    int P = in_sizes[5];

    char* ws = (char*)d_ws;
    size_t off = 0;
    auto alloc = [&](size_t bytes) -> void* {
        void* p = ws + off;
        off = (off + bytes + 255) & ~(size_t)255;
        return p;
    };
    float* emb  = (float*)alloc((size_t)N * 128 * 4);
    float* agg  = (float*)alloc((size_t)N * 128 * 4);
    float* pw   = (float*)alloc((size_t)N * 4);
    float* gsum = (float*)alloc(128 * 4);
    unsigned short* W1p = (unsigned short*)alloc(8 * 2 * 32 * 16 * 2);
    unsigned short* W2p = (unsigned short*)alloc(8 * 4 * 32 * 16 * 2);
    unsigned short* M1p = (unsigned short*)alloc(8 * 4 * 32 * 16 * 2);
    unsigned short* M2p = (unsigned short*)alloc(8 * 4 * 32 * 16 * 2);

    hipMemsetAsync(agg, 0, (size_t)N * 128 * 4, stream);
    hipMemsetAsync(pw, 0, (size_t)N * 4, stream);
    hipMemsetAsync(gsum, 0, 128 * 4, stream);

    pack_b_kernel<<<(8 * 2 * 32 + 127) / 128, 128, 0, stream>>>(W1, W1p,  38, 128, 2);
    pack_b_kernel<<<(8 * 4 * 32 + 127) / 128, 128, 0, stream>>>(W2, W2p, 128, 128, 4);
    pack_b_kernel<<<(8 * 4 * 32 + 127) / 128, 128, 0, stream>>>(M1, M1p, 128, 128, 4);
    pack_b_kernel<<<(8 * 4 * 32 + 127) / 128, 128, 0, stream>>>(M2, M2p, 128, 128, 4);

    int tiles = (N + 15) / 16;
    emb_kernel<<<tiles, 32, 0, stream>>>(bfeat, rid, remb, iemb,
                                         W1p, b1, W2p, b2, emb, N);

    pw_kernel<<<(P + 255) / 256, 256, 0, stream>>>(p2b_blk, p2b_w, pw, P);

    int ethreads_blocks = (int)(((long long)E * 32 + 255) / 256);
    b2b_kernel<<<ethreads_blocks, 256, 0, stream>>>(b2b_src, b2b_dst, b2b_w,
                                                    emb, agg, E);

    msg_kernel<<<tiles, 32, 0, stream>>>(emb, agg, pw, M1p, mb1, M2p, mb2,
                                         ln_g, ln_b, (float*)d_out, gsum, N);

    fin_kernel<<<1, 128, 0, stream>>>(gsum, (float*)d_out, N);
}
